// TeluCodec_28759101014151
// MI455X (gfx1250) — matrix operand order repaired and checked by reading, not compiled, not run on hardware
//
#include <hip/hip_runtime.h>
#include <hip/hip_bf16.h>

typedef __attribute__((ext_vector_type(16))) _Float16 v16h;
typedef __attribute__((ext_vector_type(8)))  _Float16 v8h;
typedef __attribute__((ext_vector_type(8)))  float    v8f;

#define TM 64
#define TN 32

// Build a v16h A-fragment from two contiguous 8xf16 runs (ds_load_b128 x2).
__device__ __forceinline__ v16h frag_from2(const _Float16* p0, const _Float16* p1) {
    v8h lo = *(const v8h*)p0;
    v8h hi = *(const v8h*)p1;
    return __builtin_shufflevector(lo, hi, 0, 1, 2, 3, 4, 5, 6, 7,
                                   8, 9, 10, 11, 12, 13, 14, 15);
}
// Build a v16h B-fragment from one contiguous 16xf16 run (ds_load_b128 x2).
__device__ __forceinline__ v16h frag_from16(const _Float16* p) {
    return frag_from2(p, p + 8);
}

// ---------------------------------------------------------------------------
// Generic implicit-im2col GEMM conv using WMMA f16->f32 (K>1 / strided / dilated).
//   out[b, co, n] = bias[co] + sum_{ci,kk} W(co,ci,kk) * X(b, ci, src(n,kk))
//   wmode 0: W = w[co][ci][kk]  (OIH)
//   wmode 1: W = w[ci][co][K-1-kk]  (flipped IOH, for conv_transpose)
// B tile kept n-major / k-contiguous in LDS so fragments load as b128.
// ---------------------------------------------------------------------------
__global__ __launch_bounds__(256)
void conv_gemm_wmma(const float* __restrict__ x, const float* __restrict__ w,
                    const float* __restrict__ bias, float* __restrict__ out,
                    int Ci, int Co, int Tin, int Tout,
                    int K, int stride, int pad_left, int dil, int wmode)
{
    __shared__ __align__(16) _Float16 As[TM][40];   // [m][k]
    __shared__ __align__(16) _Float16 BsT[TN][40];  // [n][k]
    const int b    = blockIdx.z;
    const int m0   = blockIdx.x * TM;
    const int n0   = blockIdx.y * TN;
    const int tid  = threadIdx.x;
    const int lane = tid & 31;
    const int wave = tid >> 5;
    const int wm   = wave & 3;      // 4 waves down M
    const int wn   = wave >> 2;     // 2 waves across N
    const int g    = lane >> 4;
    const int r    = lane & 15;
    const int Kg   = Ci * K;
    const float* xb = x + (size_t)b * Ci * Tin;
    v8f acc = {};

    for (int k0 = 0; k0 < Kg; k0 += 32) {
        // load A tile (TM x 32): ki is constant per thread -> one div per pass
        {
            const int ki = tid & 31;
            const int kg = k0 + ki;
            int ci = kg / K, kk = kg - ci * K;
            const bool kok = (kg < Kg);
            for (int mi = tid >> 5; mi < TM; mi += 8) {
                int m = m0 + mi;
                float v = 0.f;
                if (m < Co && kok) {
                    v = (wmode == 0) ? w[((size_t)m * Ci + ci) * K + kk]
                                     : w[((size_t)ci * Co + m) * K + (K - 1 - kk)];
                }
                As[mi][ki] = (_Float16)v;
            }
        }
        // load B tile (32k x TN n) with implicit im2col gather (coalesced over n)
        {
            const int ni = tid & 31;
            const int n  = n0 + ni;
            for (int ki = tid >> 5; ki < 32; ki += 8) {
                int kg = k0 + ki;
                float v = 0.f;
                if (kg < Kg && n < Tout) {
                    int ci = kg / K, kk = kg - ci * K;
                    int src = n * stride + kk - pad_left;
                    if (dil == 1) {
                        if (src >= 0 && src < Tin) v = xb[(size_t)ci * Tin + src];
                    } else if (src >= 0) {
                        int q = src / dil;
                        if (src - q * dil == 0 && q < Tin) v = xb[(size_t)ci * Tin + q];
                    }
                }
                BsT[ni][ki] = (_Float16)v;
            }
        }
        __syncthreads();
        v16h a  = frag_from2(&As[wm * 16 + r][g * 16], &As[wm * 16 + r][g * 16 + 8]);
        v16h bf = frag_from16(&BsT[wn * 16 + r][g * 16]);
        acc = __builtin_amdgcn_wmma_f32_16x16x32_f16(false, a, false, bf,
                                                     (short)0, acc, false, false);
        __syncthreads();
    }

    int n = n0 + wn * 16 + r;
    if (n < Tout) {
#pragma unroll
        for (int v = 0; v < 8; ++v) {
            int m = m0 + wm * 16 + g * 8 + v;
            if (m < Co)
                out[((size_t)b * Co + m) * Tout + n] = acc[v] + (bias ? bias[m] : 0.f);
        }
    }
}

// ---------------------------------------------------------------------------
// Fast pointwise GEMM (K=1, stride=1, pad=0, dil=1):
//   out[b,m,n] = bias[m] + sum_k w[m,k] * x[b,k,n]
// 64x64 output tile, 8 waves (4 down M x 2 across N), 2 WMMA accumulators/wave.
// Coalesced tile loads, b128 fragment loads, global_prefetch of next K tile.
// ---------------------------------------------------------------------------
__global__ __launch_bounds__(256)
void gemm_k1_wmma(const float* __restrict__ x, const float* __restrict__ w,
                  const float* __restrict__ bias, float* __restrict__ out,
                  int Ci, int Co, int T)
{
    __shared__ __align__(16) _Float16 As[64][40];   // [m][k]
    __shared__ __align__(16) _Float16 BsT[64][40];  // [n][k]
    const int b    = blockIdx.z;
    const int m0   = blockIdx.x * 64;
    const int n0   = blockIdx.y * 64;
    const int tid  = threadIdx.x;
    const int lane = tid & 31;
    const int wave = tid >> 5;
    const int wm   = wave & 3;      // 16-row M subtile
    const int wn   = wave >> 2;     // 32-col N half
    const int g    = lane >> 4;
    const int r    = lane & 15;
    const float* xb = x + (size_t)b * Ci * T;
    v8f acc0 = {}, acc1 = {};

    for (int k0 = 0; k0 < Ci; k0 += 32) {
        // prefetch next tiles into cache while this one is loaded/consumed
        if (k0 + 32 < Ci) {
            int mi = tid >> 5, ki = tid & 31;
            if (m0 + mi < Co)
                __builtin_prefetch(&w[(size_t)(m0 + mi) * Ci + k0 + 32 + ki], 0, 1);
            int ni = tid & 63, kb = tid >> 6;
            if (n0 + ni < T)
                __builtin_prefetch(&xb[(size_t)(k0 + 32 + kb) * T + n0 + ni], 0, 1);
        }
        // A tile 64x32 (coalesced along k)
        for (int i = tid; i < 64 * 32; i += 256) {
            int mi = i >> 5, ki = i & 31;
            int m = m0 + mi, k = k0 + ki;
            As[mi][ki] = (_Float16)((m < Co && k < Ci) ? w[(size_t)m * Ci + k] : 0.f);
        }
        // B tile 32k x 64n (coalesced along n; stored n-major)
        for (int i = tid; i < 32 * 64; i += 256) {
            int ki = i >> 6, ni = i & 63;
            int k = k0 + ki, n = n0 + ni;
            BsT[ni][ki] = (_Float16)((k < Ci && n < T) ? xb[(size_t)k * T + n] : 0.f);
        }
        __syncthreads();
        v16h a   = frag_from2(&As[wm * 16 + r][g * 16], &As[wm * 16 + r][g * 16 + 8]);
        v16h bf0 = frag_from16(&BsT[wn * 32 + r][g * 16]);
        v16h bf1 = frag_from16(&BsT[wn * 32 + 16 + r][g * 16]);
        acc0 = __builtin_amdgcn_wmma_f32_16x16x32_f16(false, a, false, bf0,
                                                      (short)0, acc0, false, false);
        acc1 = __builtin_amdgcn_wmma_f32_16x16x32_f16(false, a, false, bf1,
                                                      (short)0, acc1, false, false);
        __syncthreads();
    }

    int n_a = n0 + wn * 32 + r;
    int n_b = n_a + 16;
#pragma unroll
    for (int v = 0; v < 8; ++v) {
        int m = m0 + wm * 16 + g * 8 + v;
        if (m < Co) {
            float bv = bias ? bias[m] : 0.f;
            size_t rowoff = ((size_t)b * Co + m) * T;
            if (n_a < T) out[rowoff + n_a] = acc0[v] + bv;
            if (n_b < T) out[rowoff + n_b] = acc1[v] + bv;
        }
    }
}

// ---------------------------------------------------------------------------
// Persistent single-workgroup GRU. xp: (B,3H,T) with bih already folded in.
// whh16: f16 (3H,H). out: (B,H,T). H=1024, 3H=3072.
// A-side fragments load as global b128 pairs from k-contiguous whh16 rows.
// ---------------------------------------------------------------------------
__global__ __launch_bounds__(256)
void gru_kernel(const float* __restrict__ xp, const _Float16* __restrict__ whh16,
                const float* __restrict__ bhh, float* __restrict__ out,
                int B, int H, int T)
{
    __shared__ float    hf[4 * 1024];        // h (f32)  [b*H + h]
    __shared__ _Float16 h16c[1024 * 4];      // h (f16)  [k*4 + b]
    __shared__ _Float16 hg[3072 * 4];        // gates    [g3*4 + b]
    const int tid  = threadIdx.x;
    const int lane = tid & 31;
    const int wave = tid >> 5;
    const int g    = lane >> 4;
    const int r    = lane & 15;
    const int H3   = 3 * H;

    for (int i = tid; i < B * H; i += 256) { hf[i] = 0.f; }
    for (int i = tid; i < H * 4; i += 256) { h16c[i] = (_Float16)0.f; }
    __syncthreads();

    for (int t = 0; t < T; ++t) {
        for (int tile = wave; tile < H3 / 16; tile += 8) {
            int mbase = tile * 16;
            const _Float16* wrow = whh16 + (size_t)(mbase + r) * H;
            v8f acc = {};
            for (int k0 = 0; k0 < H; k0 += 32) {
                v16h a = frag_from2(wrow + k0 + g * 16, wrow + k0 + g * 16 + 8);
                v16h bf;
#pragma unroll
                for (int j = 0; j < 16; ++j) {
                    int kB = k0 + g * 16 + j;
                    bf[j] = (r < 4) ? h16c[kB * 4 + r] : (_Float16)0.f;
                }
                acc = __builtin_amdgcn_wmma_f32_16x16x32_f16(false, a, false, bf,
                                                             (short)0, acc, false, false);
            }
            if (r < 4) {
#pragma unroll
                for (int v = 0; v < 8; ++v)
                    hg[(mbase + g * 8 + v) * 4 + r] = (_Float16)acc[v];
            }
        }
        __syncthreads();
        for (int i = tid; i < B * H; i += 256) {
            int b  = i / H;
            int hh = i - b * H;
            size_t xbase = ((size_t)b * H3) * T + t;
            float xr = xp[xbase + (size_t)hh * T];
            float xz = xp[xbase + (size_t)(H + hh) * T];
            float xn = xp[xbase + (size_t)(2 * H + hh) * T];
            float hr = (float)hg[hh * 4 + b]           + bhh[hh];
            float hz = (float)hg[(H + hh) * 4 + b]     + bhh[H + hh];
            float hn = (float)hg[(2 * H + hh) * 4 + b] + bhh[2 * H + hh];
            float rg = 1.f / (1.f + __expf(-(xr + hr)));
            float zg = 1.f / (1.f + __expf(-(xz + hz)));
            float ng = tanhf(xn + rg * hn);
            float hprev = hf[i];
            float hnew  = (1.f - zg) * ng + zg * hprev;
            hf[i] = hnew;
            h16c[hh * 4 + b] = (_Float16)hnew;
            out[((size_t)b * H + hh) * T + t] = hnew;
        }
        __syncthreads();
    }
}

// ----------------------- elementwise / reduction helpers --------------------
__global__ void f2h_kernel(const float* __restrict__ x, _Float16* __restrict__ y, long n) {
    for (long i = blockIdx.x * (long)blockDim.x + threadIdx.x; i < n;
         i += (long)gridDim.x * blockDim.x) y[i] = (_Float16)x[i];
}
__global__ void zero_kernel(float* __restrict__ x, long n) {
    for (long i = blockIdx.x * (long)blockDim.x + threadIdx.x; i < n;
         i += (long)gridDim.x * blockDim.x) x[i] = 0.f;
}
__device__ float block_reduce_sum(float v) {
    __shared__ float sh[256];
    __syncthreads();
    sh[threadIdx.x] = v;
    __syncthreads();
    for (int s = 128; s > 0; s >>= 1) {
        if ((int)threadIdx.x < s) sh[threadIdx.x] += sh[threadIdx.x + s];
        __syncthreads();
    }
    return sh[0];
}
__global__ __launch_bounds__(256)
void gn_stats(const float* __restrict__ x, float* __restrict__ mean,
              float* __restrict__ var, long CT) {
    int b = blockIdx.x;
    const float* xb = x + (size_t)b * CT;
    float s = 0.f, ss = 0.f;
    for (long i = threadIdx.x; i < CT; i += blockDim.x) { float v = xb[i]; s += v; ss += v * v; }
    float S = block_reduce_sum(s);
    float SS = block_reduce_sum(ss);
    if (threadIdx.x == 0) { float m = S / (float)CT; mean[b] = m; var[b] = SS / (float)CT - m * m; }
}
__global__ void gn_gelu_apply(const float* __restrict__ x, float* __restrict__ y,
                              const float* __restrict__ ga, const float* __restrict__ be,
                              const float* __restrict__ mean, const float* __restrict__ var,
                              int C, int T, long n) {
    for (long i = blockIdx.x * (long)blockDim.x + threadIdx.x; i < n;
         i += (long)gridDim.x * blockDim.x) {
        long ct = (long)C * T;
        int b = (int)(i / ct);
        int c = (int)((i / T) % C);
        float inv = rsqrtf(var[b] + 1e-5f);
        float h = (x[i] - mean[b]) * inv * ga[c] + be[c];
        y[i] = 0.5f * h * (1.f + erff(h * 0.70710678118654752440f));
    }
}
__global__ void add_inplace(float* __restrict__ x, const float* __restrict__ y, long n) {
    for (long i = blockIdx.x * (long)blockDim.x + threadIdx.x; i < n;
         i += (long)gridDim.x * blockDim.x) x[i] += y[i];
}
__global__ void bn_tanh_kernel(float* __restrict__ x, const float* __restrict__ m,
                               const float* __restrict__ v, const float* __restrict__ ga,
                               const float* __restrict__ be, int C, int T, long n) {
    for (long i = blockIdx.x * (long)blockDim.x + threadIdx.x; i < n;
         i += (long)gridDim.x * blockDim.x) {
        int c = (int)((i / T) % C);
        float h = (x[i] - m[c]) * rsqrtf(v[c] + 1e-5f) * ga[c] + be[c];
        x[i] = tanhf(h);
    }
}
__global__ void tanh_inplace(float* __restrict__ x, long n) {
    for (long i = blockIdx.x * (long)blockDim.x + threadIdx.x; i < n;
         i += (long)gridDim.x * blockDim.x) x[i] = tanhf(x[i]);
}
// --------------------------------- VQ ---------------------------------------
__global__ void cbsq_kernel(const float* __restrict__ cb, float* __restrict__ cbsq,
                            int Ncb, int D) {
    int j = blockIdx.x * blockDim.x + threadIdx.x;
    if (j >= Ncb) return;
    float s = 0.f;
    const float* row = cb + (size_t)j * D;
    for (int d = 0; d < D; ++d) { float v = row[d]; s += v * v; }
    cbsq[j] = s;
}
__global__ void vq_argmin(const float* __restrict__ dot, const float* __restrict__ cbsq,
                          int* __restrict__ codes, int qidx, int B, int T, int Ncb) {
    int i = blockIdx.x * blockDim.x + threadIdx.x;
    if (i >= B * T) return;
    int b = i / T, t = i - b * T;
    const float* dp = dot + ((size_t)b * Ncb) * T + t;
    float best = cbsq[0] - 2.f * dp[0];
    int bi = 0;
    for (int j = 1; j < Ncb; ++j) {
        float s = cbsq[j] - 2.f * dp[(size_t)j * T];
        if (s < best) { best = s; bi = j; }
    }
    codes[((size_t)b * 8 + qidx) * T + t] = bi;
}
__global__ void vq_update(float* __restrict__ rT, float* __restrict__ qz,
                          const float* __restrict__ cb, const int* __restrict__ codes,
                          int qidx, int B, int D, int T) {
    long n = (long)B * D * T;
    for (long i = blockIdx.x * (long)blockDim.x + threadIdx.x; i < n;
         i += (long)gridDim.x * blockDim.x) {
        long dt = (long)D * T;
        int b = (int)(i / dt);
        long rem = i - (long)b * dt;
        int d = (int)(rem / T);
        int t = (int)(rem - (long)d * T);
        int code = codes[((size_t)b * 8 + qidx) * T + t];
        float v = cb[(size_t)code * D + d];
        qz[i] += v;
        rT[i] -= v;
    }
}
// ------------------------------ reductions -----------------------------------
__global__ __launch_bounds__(256)
void reduce_sumsq(const float* __restrict__ x, float* __restrict__ slot, long n) {
    float s = 0.f;
    for (long i = threadIdx.x; i < n; i += blockDim.x) { float v = x[i]; s += v * v; }
    float S = block_reduce_sum(s);
    if (threadIdx.x == 0) *slot = S;
}
__global__ __launch_bounds__(256)
void reduce_sumsq_batch(const float* __restrict__ x, float* __restrict__ slots, long n) {
    int b = blockIdx.x;
    const float* xb = x + (size_t)b * n;
    float s = 0.f;
    for (long i = threadIdx.x; i < n; i += blockDim.x) { float v = xb[i]; s += v * v; }
    float S = block_reduce_sum(s);
    if (threadIdx.x == 0) slots[b] = S;
}
__global__ __launch_bounds__(256)
void reduce_absdiff(const float* __restrict__ a, const float* __restrict__ b,
                    float* __restrict__ slot, long n) {
    float s = 0.f;
    for (long i = threadIdx.x; i < n; i += blockDim.x) s += fabsf(a[i] - b[i]);
    float S = block_reduce_sum(s);
    if (threadIdx.x == 0) *slot = S;
}
// ------------------------------ outputs --------------------------------------
__global__ void write_recon(const float* __restrict__ rec, float* __restrict__ out,
                            int B, int Tr, int L) {
    long n = (long)B * L;
    for (long i = blockIdx.x * (long)blockDim.x + threadIdx.x; i < n;
         i += (long)gridDim.x * blockDim.x) {
        int b = (int)(i / L);
        int t = (int)(i - (long)b * L);
        out[i] = (t < Tr) ? rec[(size_t)b * Tr + t] : 0.f;
    }
}
__global__ void write_codes(const int* __restrict__ codes, float* __restrict__ out, long n) {
    for (long i = blockIdx.x * (long)blockDim.x + threadIdx.x; i < n;
         i += (long)gridDim.x * blockDim.x) out[i] = (float)codes[i];
}
__global__ void final_losses(const float* __restrict__ SC, float* __restrict__ out_total,
                             long btd, long nl, int B, long L) {
    if (threadIdx.x != 0 || blockIdx.x != 0) return;
    float vq = 0.f;
    for (int q = 0; q < 8; ++q) {
        float cl = SC[q] / (float)btd;
        cl = fminf(fmaxf(cl, 0.f), 10.f);
        vq += 0.25f * cl;
    }
    vq = fminf(fmaxf(vq, 0.f), 10.f);
    float rl = SC[8] / (float)nl;
    rl = fminf(fmaxf(rl, 0.f), 10.f);
    float sl = 0.f;
    for (int b = 0; b < B; ++b) {
        float irms = sqrtf(SC[9 + b] / (float)L + 1e-8f);
        float orms = sqrtf(SC[13 + b] / (float)L + 1e-8f);
        float d = orms - irms;
        sl += d * d;
    }
    sl = sl / (float)B * 10.f;
    *out_total = rl + vq + sl;
}

// ============================================================================
// Host orchestration
// ============================================================================
static inline int ew_blocks(long n) {
    long b = (n + 255) / 256;
    if (b > 4096) b = 4096;
    if (b < 1) b = 1;
    return (int)b;
}

static void launch_conv(const float* x, const float* w, const float* bias, float* out,
                        int B, int Ci, int Co, int Tin, int Tout, int K, int stride,
                        int padl, int dil, int wmode, hipStream_t s) {
    if (K == 1 && stride == 1 && dil == 1 && padl == 0 && wmode == 0) {
        dim3 grid((Co + 63) / 64, (Tout + 63) / 64, B);
        gemm_k1_wmma<<<grid, 256, 0, s>>>(x, w, bias, out, Ci, Co, Tout);
    } else {
        dim3 grid((Co + TM - 1) / TM, (Tout + TN - 1) / TN, B);
        conv_gemm_wmma<<<grid, 256, 0, s>>>(x, w, bias, out, Ci, Co, Tin, Tout,
                                            K, stride, padl, dil, wmode);
    }
}

struct Ws {
    float* SC;
    float* A; float* Bb; float* Cc;
    float* XP; float* QZ; float* DOT; float* CBSQ;
    int*   CODES;
    _Float16* WHH16;
};

static void res_block(float* x, float* t1, float* t2, void* const* d_in, int p,
                      int B, int C, int T, float* SC, hipStream_t s) {
    long n = (long)B * C * T;
    gn_stats<<<B, 256, 0, s>>>(x, SC + 20, SC + 24, (long)C * T);
    gn_gelu_apply<<<ew_blocks(n), 256, 0, s>>>(x, t1, (const float*)d_in[p + 0],
                                               (const float*)d_in[p + 1], SC + 20, SC + 24, C, T, n);
    launch_conv(t1, (const float*)d_in[p + 2], (const float*)d_in[p + 3], t2,
                B, C, C, T, T, 3, 1, 2, 1, 0, s);
    gn_stats<<<B, 256, 0, s>>>(t2, SC + 20, SC + 24, (long)C * T);
    gn_gelu_apply<<<ew_blocks(n), 256, 0, s>>>(t2, t1, (const float*)d_in[p + 4],
                                               (const float*)d_in[p + 5], SC + 20, SC + 24, C, T, n);
    launch_conv(t1, (const float*)d_in[p + 6], (const float*)d_in[p + 7], t2,
                B, C, C, T, T, 1, 1, 0, 1, 0, s);
    add_inplace<<<ew_blocks(n), 256, 0, s>>>(x, t2, n);
}

extern "C" void kernel_launch(void* const* d_in, const int* in_sizes, int n_in,
                              void* d_out, int out_size, void* d_ws, size_t ws_size,
                              hipStream_t stream) {
    if (n_in < 138) return;
    const int B = 4, H = 1024, L = 64000;

    float* ws = (float*)d_ws;
    size_t off = 0;
    auto alloc = [&](size_t nf) { float* p = ws + off; off += (nf + 63) & ~(size_t)63; return p; };
    Ws W;
    W.SC    = alloc(64);
    W.A     = alloc(8500000);
    W.Bb    = alloc(8500000);
    W.Cc    = alloc(8500000);
    W.XP    = alloc(9900000);
    W.QZ    = alloc(3300000);
    W.DOT   = alloc(3300000);
    W.CBSQ  = alloc(1024);
    W.CODES = (int*)alloc(25600);
    W.WHH16 = (_Float16*)alloc(3072 * 1024 / 2 + 64);

    const float* audio = (const float*)d_in[0];

    zero_kernel<<<1, 64, 0, stream>>>(W.SC, 64);

    // ---- encoder ----
    auto causal_out = [](int T, int k, int s) { return (T + k - 2) / s + 1 - (k - 1); };
    const int eK[6] = {7, 7, 7, 7, 7, 3};
    const int eS[6] = {1, 2, 2, 2, 5, 2};
    const int eC[7] = {1, 32, 64, 128, 256, 512, 1024};
    int T = L;
    float* cur = nullptr;
    const float* src = audio;
    float* bufs[2] = {W.A, W.Bb};
    for (int i = 0; i < 6; ++i) {
        int Tout = causal_out(T, eK[i], eS[i]);
        float* dst = bufs[i & 1];
        launch_conv(src, (const float*)d_in[1 + 2 * i], (const float*)d_in[2 + 2 * i], dst,
                    B, eC[i], eC[i + 1], T, Tout, eK[i], eS[i], eK[i] - 1, 1, 0, stream);
        float* o1 = bufs[(i + 1) & 1];
        res_block(dst, o1, W.Cc, d_in, 13 + 8 * i, B, eC[i + 1], Tout, W.SC, stream);
        cur = dst; src = dst; T = Tout;
    }
    const int Tz = T;  // 797
    float* zbuf = (cur == W.A) ? W.Bb : W.A;
    launch_conv(cur, (const float*)d_in[61], (const float*)d_in[62], zbuf,
                B, H, H, Tz, Tz, 1, 1, 0, 1, 0, stream);

    // ---- GRU ----
    f2h_kernel<<<ew_blocks(3072 * 1024), 256, 0, stream>>>((const float*)d_in[64], W.WHH16,
                                                           (long)3072 * 1024);
    launch_conv(zbuf, (const float*)d_in[63], (const float*)d_in[65], W.XP,
                B, H, 3 * H, Tz, Tz, 1, 1, 0, 1, 0, stream);
    float* RT = W.Cc;
    gru_kernel<<<1, 256, 0, stream>>>(W.XP, W.WHH16, (const float*)d_in[66], RT, B, H, Tz);

    // ---- residual VQ ----
    long nzd = (long)B * H * Tz;
    zero_kernel<<<ew_blocks(nzd), 256, 0, stream>>>(W.QZ, nzd);
    const float* cbs = (const float*)d_in[67];
    for (int q = 0; q < 8; ++q) {
        const float* cb = cbs + (size_t)q * 1024 * H;
        cbsq_kernel<<<(1024 + 255) / 256, 256, 0, stream>>>(cb, W.CBSQ, 1024, H);
        launch_conv(RT, cb, nullptr, W.DOT, B, H, 1024, Tz, Tz, 1, 1, 0, 1, 0, stream);
        vq_argmin<<<(B * Tz + 255) / 256, 256, 0, stream>>>(W.DOT, W.CBSQ, W.CODES, q, B, Tz, 1024);
        vq_update<<<ew_blocks(nzd), 256, 0, stream>>>(RT, W.QZ, cb, W.CODES, q, B, H, Tz);
        reduce_sumsq<<<1, 256, 0, stream>>>(RT, W.SC + q, nzd);
    }

    // ---- decoder ----
    launch_conv(W.QZ, (const float*)d_in[68], (const float*)d_in[69], W.A,
                B, H, H, Tz, Tz, 1, 1, 0, 1, 0, stream);
    res_block(W.A, W.Bb, W.Cc, d_in, 70, B, H, Tz, W.SC, stream);

    const int tS[5] = {2, 5, 2, 2, 2};
    const int tP[5] = {1, 2, 1, 1, 1};
    const int tK[5] = {4, 10, 4, 4, 4};
    const int dC[6] = {1024, 512, 256, 128, 64, 32};
    auto tconv_out = [](int Tin, int s, int k, int p) {
        return (Tin - 1) * s + 1 + 2 * (k - 1 - p) - k + 1;
    };
    T = Tz;
    cur = W.A;
    for (int i = 0; i < 5; ++i) {
        int Tout = tconv_out(T, tS[i], tK[i], tP[i]);
        float* dst = (cur == W.A) ? W.Bb : W.A;
        launch_conv(cur, (const float*)d_in[118 + 2 * i], (const float*)d_in[119 + 2 * i], dst,
                    B, dC[i], dC[i + 1], T, Tout, tK[i], 1, tK[i] - 1 - tP[i], tS[i], 1, stream);
        float* o1 = (dst == W.A) ? W.Bb : W.A;
        res_block(dst, o1, W.Cc, d_in, 70 + 8 * (i + 1), B, dC[i + 1], Tout, W.SC, stream);
        cur = dst; T = Tout;
    }
    const int Tr = T;  // 63768
    float* ybuf = (cur == W.A) ? W.Bb : W.A;
    launch_conv(cur, (const float*)d_in[128], (const float*)d_in[129], ybuf,
                B, 32, 1, Tr, Tr, 7, 1, 3, 1, 0, stream);

    // ---- postnet ----
    float* hbuf = (ybuf == W.A) ? W.Bb : W.A;
    launch_conv(ybuf, (const float*)d_in[130], (const float*)d_in[131], hbuf,
                B, 1, 16, Tr, Tr, 5, 1, 2, 1, 0, stream);
    long nh = (long)B * 16 * Tr;
    bn_tanh_kernel<<<ew_blocks(nh), 256, 0, stream>>>(hbuf, (const float*)d_in[134],
                                                      (const float*)d_in[135],
                                                      (const float*)d_in[132],
                                                      (const float*)d_in[133], 16, Tr, nh);
    float* recon = W.Cc;
    launch_conv(hbuf, (const float*)d_in[136], (const float*)d_in[137], recon,
                B, 16, 1, Tr, Tr, 5, 1, 2, 1, 0, stream);
    long nr = (long)B * Tr;
    tanh_inplace<<<ew_blocks(nr), 256, 0, stream>>>(recon, nr);

    // ---- outputs + losses ----
    float* out = (float*)d_out;
    write_recon<<<ew_blocks((long)B * L), 256, 0, stream>>>(recon, out, B, Tr, L);
    reduce_absdiff<<<1, 256, 0, stream>>>(out, audio, W.SC + 8, (long)B * L);
    reduce_sumsq_batch<<<B, 256, 0, stream>>>(audio, W.SC + 9, (long)L);
    reduce_sumsq_batch<<<B, 256, 0, stream>>>(out, W.SC + 13, (long)L);
    long nc = (long)B * 8 * Tz;
    write_codes<<<ew_blocks(nc), 256, 0, stream>>>(W.CODES, out + (long)B * L, nc);
    final_losses<<<1, 1, 0, stream>>>(W.SC, out + (out_size - 1),
                                      (long)B * Tz * H, (long)B * L, B, (long)L);
    (void)in_sizes; (void)ws_size;
}
